// Attend_70789650973195
// MI455X (gfx1250) — matrix operand order repaired and checked by reading, not compiled, not run on hardware
//
#include <hip/hip_runtime.h>

typedef __attribute__((ext_vector_type(16))) _Float16 v16h;
typedef __attribute__((ext_vector_type(8)))  _Float16 v8h;
typedef __attribute__((ext_vector_type(8)))  float    v8f;
typedef __attribute__((ext_vector_type(4)))  unsigned int u32x4;
typedef __attribute__((ext_vector_type(8)))  int          i32x8;
typedef __attribute__((ext_vector_type(4)))  int          i32x4;

#define N_CTX 2048
#define DHEAD 64
#define BC    64     // keys per main-loop iteration
#define BR    64     // query rows per block (4 waves * 16 rows)

#define WMMA_F16(A, B, C) \
    __builtin_amdgcn_wmma_f32_16x16x32_f16(false, (A), false, (B), (short)0, (C), false, false)

// ---- TDM: 2D tile (tile1 rows x tile0 elems, f16) global -> LDS ----
static __device__ inline void tdm_load_2d(unsigned lds_addr, const void* gptr,
                                          unsigned dim0, unsigned dim1,
                                          unsigned stride0, unsigned tile0,
                                          unsigned tile1)
{
    unsigned long long ga = (unsigned long long)(size_t)gptr;
    u32x4 g0;
    g0[0] = 1u;                                   // count=1, user desc
    g0[1] = lds_addr;                             // LDS byte address
    g0[2] = (unsigned)ga;                         // global_addr[31:0]
    g0[3] = (unsigned)((ga >> 32) & 0x1FFFFFFu) | (2u << 30);  // addr[56:32], type=2
    i32x8 g1;
    g1[0] = (int)(1u << 16);                      // data_size = 2 bytes
    g1[1] = (int)((dim0 & 0xFFFFu) << 16);        // tensor_dim0[15:0]
    g1[2] = (int)(((dim0 >> 16) & 0xFFFFu) | ((dim1 & 0xFFFFu) << 16));
    g1[3] = (int)(((dim1 >> 16) & 0xFFFFu) | ((tile0 & 0xFFFFu) << 16));
    g1[4] = (int)(tile1 & 0xFFFFu);               // tile_dim1 (tile_dim2 = 0)
    g1[5] = (int)stride0;                         // tensor_dim0_stride[31:0]
    g1[6] = 0;
    g1[7] = 0;
    i32x4 z4 = {0, 0, 0, 0};
#if __clang_major__ >= 23
    i32x8 z8 = {0, 0, 0, 0, 0, 0, 0, 0};
    __builtin_amdgcn_tensor_load_to_lds(g0, g1, z4, z4, z8, 0);
#else
    __builtin_amdgcn_tensor_load_to_lds(g0, g1, z4, z4, 0);
#endif
}

// ---- prep pass: K fp32 -> f16 [bh][key][d]; V fp32 -> f16 transposed [bh][d][key]
__global__ __launch_bounds__(256) void prep_kernel(
    const float* __restrict__ k, const float* __restrict__ v,
    _Float16* __restrict__ k16, _Float16* __restrict__ v16t)
{
    __shared__ __align__(128) _Float16 sT[64 * 64];   // 8 KB V tile
    const int bh = blockIdx.y;
    const int j0 = blockIdx.x * 64;
    const float* kb = k + ((size_t)bh * N_CTX) * DHEAD;
    const float* vb = v + ((size_t)bh * N_CTX) * DHEAD;

    const int tid = threadIdx.x;
    const int row = tid >> 2;            // 0..63 key row
    const int c0  = (tid & 3) * 16;      // 0,16,32,48

    // K: straight convert, coalesced
    {
        const float4* kp = (const float4*)(kb + (size_t)(j0 + row) * DHEAD + c0);
        #pragma unroll
        for (int i = 0; i < 2; ++i) {
            float4 f0 = kp[2 * i], f1 = kp[2 * i + 1];
            v8h h;
            h[0]=(_Float16)f0.x; h[1]=(_Float16)f0.y; h[2]=(_Float16)f0.z; h[3]=(_Float16)f0.w;
            h[4]=(_Float16)f1.x; h[5]=(_Float16)f1.y; h[6]=(_Float16)f1.z; h[7]=(_Float16)f1.w;
            *(v8h*)&k16[((size_t)bh * N_CTX + j0 + row) * DHEAD + c0 + i * 8] = h;
        }
    }
    // V: convert into LDS tile, then write transposed (coalesced 32B stores)
    {
        const float4* vp = (const float4*)(vb + (size_t)(j0 + row) * DHEAD + c0);
        #pragma unroll
        for (int i = 0; i < 2; ++i) {
            float4 f0 = vp[2 * i], f1 = vp[2 * i + 1];
            v8h h;
            h[0]=(_Float16)f0.x; h[1]=(_Float16)f0.y; h[2]=(_Float16)f0.z; h[3]=(_Float16)f0.w;
            h[4]=(_Float16)f1.x; h[5]=(_Float16)f1.y; h[6]=(_Float16)f1.z; h[7]=(_Float16)f1.w;
            *(v8h*)&sT[(size_t)row * 64 + c0 + i * 8] = h;
        }
    }
    __syncthreads();
    {
        const int drow = tid >> 2;       // d row 0..63
        const int k0   = (tid & 3) * 16; // key group
        v16h t;
        #pragma unroll
        for (int i = 0; i < 16; ++i) t[i] = sT[(size_t)(k0 + i) * 64 + drow];
        *(v16h*)&v16t[((size_t)bh * DHEAD + drow) * N_CTX + j0 + k0] = t;
    }
}

// ---- main flash-attention kernel: TDM-staged f16 tiles, f16 WMMA compute ----
__global__ __launch_bounds__(128) void fattn_kernel(
    const float* __restrict__ q, const _Float16* __restrict__ k16,
    const _Float16* __restrict__ v16t, float* __restrict__ out)
{
    __shared__ __align__(128) _Float16 sK [2][BC * DHEAD];   // 16 KB [key][d]
    __shared__ __align__(128) _Float16 sVt[2][DHEAD * BC];   // 16 KB [d][key]
    __shared__ __align__(128) _Float16 sP [4][16 * BC];      //  8 KB per-wave P

    const int lane = threadIdx.x & 31;
    const int wave = threadIdx.x >> 5;
    const int lrow = lane & 15;
    const int lhi  = lane >> 4;

    const int bh = blockIdx.y;
    const float*    qb  = q    + ((size_t)bh * N_CTX) * DHEAD;
    const _Float16* kbh = k16  + ((size_t)bh * N_CTX) * DHEAD;
    const _Float16* vbh = v16t + ((size_t)bh * DHEAD) * N_CTX;
    float*          ob  = out  + ((size_t)bh * N_CTX) * DHEAD;

    const int q0 = blockIdx.x * BR + wave * 16;

    // ---- Q fragments (A layout), scale folded in ----
    v16h a0, a1;
    const float scale = 0.125f;
    #pragma unroll
    for (int e = 0; e < 16; ++e) {
        int c = (lhi << 4) + e;
        a0[e] = (_Float16)(qb[(size_t)(q0 + lrow) * DHEAD + c] * scale);
        a1[e] = (_Float16)(qb[(size_t)(q0 + lrow) * DHEAD + 32 + c] * scale);
    }

    v16h ones;
    #pragma unroll
    for (int e = 0; e < 16; ++e) ones[e] = (_Float16)1.0f;

    v8f oacc[4] = {v8f{}, v8f{}, v8f{}, v8f{}};
    v8f lacc = {};
    float m[8];
    #pragma unroll
    for (int r = 0; r < 8; ++r) m[r] = -1e30f;

    // ---- TDM prologue: stage tile 0 into buffer 0 ----
    if (threadIdx.x < 32) {
        tdm_load_2d((unsigned)(size_t)&sK[0][0],  kbh, DHEAD, N_CTX, DHEAD, DHEAD, BC);
        tdm_load_2d((unsigned)(size_t)&sVt[0][0], vbh, N_CTX, DHEAD, N_CTX, BC, DHEAD);
    }

    for (int j0 = 0; j0 < N_CTX; j0 += BC) {
        const int sj = (j0 / BC) & 1;
        __syncthreads();   // all readers of buffer sj^1 (tile j-1) are done
        if (threadIdx.x < 32) {
            const int jn = j0 + BC;
            if (jn < N_CTX) {   // prefetch tile j+1 into the other buffer
                tdm_load_2d((unsigned)(size_t)&sK[sj ^ 1][0],
                            kbh + (size_t)jn * DHEAD,
                            DHEAD, N_CTX - jn, DHEAD, DHEAD, BC);
                tdm_load_2d((unsigned)(size_t)&sVt[sj ^ 1][0],
                            vbh + jn,
                            N_CTX - jn, DHEAD, N_CTX, BC, DHEAD);
                __builtin_amdgcn_s_wait_tensorcnt(2);  // tile j resident (in-order)
            } else {
                __builtin_amdgcn_s_wait_tensorcnt(0);
            }
        }
        __syncthreads();   // release: tile j is in LDS

        const _Float16* cK  = &sK[sj][0];
        const _Float16* cVt = &sVt[sj][0];

        // ---- S = Q*K^T : four 16x16 key tiles, contraction 64 ----
        v8f s[4] = {v8f{}, v8f{}, v8f{}, v8f{}};
        #pragma unroll
        for (int kc = 0; kc < 2; ++kc) {
            v16h aa = kc ? a1 : a0;
            #pragma unroll
            for (int tt = 0; tt < 4; ++tt) {
                v16h bk = *(const v16h*)&cK[(size_t)(tt * 16 + lrow) * DHEAD + kc * 32 + lhi * 16];
                s[tt] = WMMA_F16(aa, bk, s[tt]);
            }
        }

        // ---- online softmax: row max butterfly, P to LDS ----
        float alpha[8];
        #pragma unroll
        for (int r = 0; r < 8; ++r) {
            float tmax = fmaxf(fmaxf(s[0][r], s[1][r]), fmaxf(s[2][r], s[3][r]));
            #pragma unroll
            for (int off = 1; off < 16; off <<= 1)
                tmax = fmaxf(tmax, __shfl_xor(tmax, off, 32));
            float mnew = fmaxf(m[r], tmax);
            alpha[r] = __expf(m[r] - mnew);
            m[r] = mnew;
            int prow = r + lhi * 8;
            #pragma unroll
            for (int tt = 0; tt < 4; ++tt) {
                float p = __expf(s[tt][r] - mnew);
                sP[wave][prow * BC + tt * 16 + lrow] = (_Float16)p;
            }
        }
        asm volatile("s_wait_dscnt 0" ::: "memory");   // intra-wave P transpose

        // ---- P as two A fragments ----
        const _Float16* sPw = &sP[wave][0];
        v8h p0lo = *(const v8h*)&sPw[lrow * BC + lhi * 16];
        v8h p0hi = *(const v8h*)&sPw[lrow * BC + 8 + lhi * 16];
        v8h p1lo = *(const v8h*)&sPw[lrow * BC + 32 + lhi * 16];
        v8h p1hi = *(const v8h*)&sPw[lrow * BC + 40 + lhi * 16];
        v16h pa0 = __builtin_shufflevector(p0lo, p0hi, 0,1,2,3,4,5,6,7,8,9,10,11,12,13,14,15);
        v16h pa1 = __builtin_shufflevector(p1lo, p1hi, 0,1,2,3,4,5,6,7,8,9,10,11,12,13,14,15);

        // ---- rescale accumulators ----
        #pragma unroll
        for (int r = 0; r < 8; ++r) {
            lacc[r] *= alpha[r];
            #pragma unroll
            for (int t = 0; t < 4; ++t) oacc[t][r] *= alpha[r];
        }

        // ---- row-sum of P on the matrix unit ----
        lacc = WMMA_F16(pa0, ones, lacc);
        lacc = WMMA_F16(pa1, ones, lacc);

        // ---- O += P * V ----
        #pragma unroll
        for (int kc = 0; kc < 2; ++kc) {
            v16h pp = kc ? pa1 : pa0;
            #pragma unroll
            for (int t = 0; t < 4; ++t) {
                v16h bv = *(const v16h*)&cVt[(size_t)(t * 16 + lrow) * BC + kc * 32 + lhi * 16];
                oacc[t] = WMMA_F16(pp, bv, oacc[t]);
            }
        }
    }

    // ---- normalize and store fp32 ----
    #pragma unroll
    for (int r = 0; r < 8; ++r) {
        float inv = 1.0f / lacc[r];
        int row = q0 + r + lhi * 8;
        #pragma unroll
        for (int t = 0; t < 4; ++t)
            ob[(size_t)row * DHEAD + t * 16 + lrow] = oacc[t][r] * inv;
    }
}

extern "C" void kernel_launch(void* const* d_in, const int* in_sizes, int n_in,
                              void* d_out, int out_size, void* d_ws, size_t ws_size,
                              hipStream_t stream) {
    const float* q = (const float*)d_in[0];
    const float* k = (const float*)d_in[1];
    const float* v = (const float*)d_in[2];
    float* out = (float*)d_out;

    const int bh = in_sizes[0] / (N_CTX * DHEAD);   // B*H = 64
    _Float16* k16  = (_Float16*)d_ws;                               // 16 MB
    _Float16* v16t = k16 + (size_t)bh * N_CTX * DHEAD;              // 16 MB

    dim3 pgrid(N_CTX / 64, bh);
    prep_kernel<<<pgrid, 256, 0, stream>>>(k, v, k16, v16t);

    dim3 grid(N_CTX / BR, bh);
    fattn_kernel<<<grid, 128, 0, stream>>>(q, k16, v16t, out);
}
